// Hopfield_44684839748140
// MI455X (gfx1250) — matrix operand order repaired and checked by reading, not compiled, not run on hardware
//
#include <hip/hip_runtime.h>
#include <hip/hip_bf16.h>
#include <stdint.h>

// ---------------------------------------------------------------------------
// CDNA5 (gfx1250) Hopfield attention: bf16 WMMA everywhere, sparsemax via
// bisection in 256KB LDS score slabs, async global->LDS double buffering.
// ---------------------------------------------------------------------------

typedef __attribute__((ext_vector_type(16))) __bf16 v16bf;
typedef __attribute__((ext_vector_type(8)))  float  v8f;

#if __has_builtin(__builtin_amdgcn_global_load_async_to_lds_b128)
#define USE_ASYNC_LDS 1
// builtin prototype: (v4i addrspace(1)*, v4i addrspace(3)*, imm offset, imm cpol)
typedef int v4i __attribute__((vector_size(16)));
typedef __attribute__((address_space(1))) v4i v4i_g;
typedef __attribute__((address_space(3))) v4i v4i_l;
#endif

__device__ __forceinline__ void wait_async_lds() {
#ifdef USE_ASYNC_LDS
#if __has_builtin(__builtin_amdgcn_s_wait_asynccnt)
  __builtin_amdgcn_s_wait_asynccnt(0);
#else
  asm volatile("s_wait_asynccnt 0x0" ::: "memory");
#endif
#endif
}

union FragU {
  v16bf    v;
  uint32_t u[8];
  uint4    q[2];
};

__device__ __forceinline__ uint16_t f2bf(float x) {
  uint32_t u = __builtin_bit_cast(uint32_t, x);
  uint32_t r = u + 0x7FFFu + ((u >> 16) & 1u);   // round-to-nearest-even
  return (uint16_t)(r >> 16);
}

// A-operand fragment (16x32 bf16, MxK): lane l holds row (l&15); K is
// contiguous per lane group g = l>>4: pair q -> dword offset q+8g.
__device__ __forceinline__ v16bf load_frag_a(const uint16_t* p, int g) {
  FragU f;
  const uint32_t* d = (const uint32_t*)p;
  int b = g * 8;
#pragma unroll
  for (int q = 0; q < 8; ++q) {
    int off = b + q;
    f.u[q] = d[off];
  }
  return f.v;
}

// B-operand fragment (32x16 bf16, KxN): lane l holds column (l&15); K is
// contiguous per lane group (lanes 0-15: K=0..15, lanes 16-31: K=16..31).
__device__ __forceinline__ v16bf load_frag_b(const uint16_t* p, int g) {
  FragU f;
  const uint4* d = (const uint4*)(p + g * 16);
  f.q[0] = d[0];
  f.q[1] = d[1];
  return f.v;
}

__device__ __forceinline__ v8f wmma_bf16(v16bf a, v16bf b, v8f c) {
  return __builtin_amdgcn_wmma_f32_16x16x32_bf16(
      /*neg_a=*/false, a, /*neg_b=*/false, b,
      /*c_mod=*/(short)0, c, /*reuse_a=*/false, /*reuse_b=*/false);
}

// ---------------------------------------------------------------------------
// fp32 -> bf16 conversion
// ---------------------------------------------------------------------------
__global__ __launch_bounds__(256) void cvt_f32_bf16(const float* __restrict__ in,
                                                    uint16_t* __restrict__ out,
                                                    int n) {
  int i = blockIdx.x * 256 + threadIdx.x;
  if (i < n) out[i] = f2bf(in[i]);
}

// ---------------------------------------------------------------------------
// Generic bf16 GEMM: C = A[MxK] * B[KxN] + bias[N]
//   MODE 0: f32 out row-major      MODE 1: bf16 out row-major
//   MODE 2: bf16 out transposed (out[n*M + m])  -- used to produce V^T
// Tile 128x64, BK=32, 256 threads (8 waves, 4x2 wave grid, 32x32 per wave).
// Double-buffered LDS; A tile staged via async global->LDS DMA when available.
// ---------------------------------------------------------------------------
#define GTM 128
#define GTN 64
#define GTK 32

template <int MODE>
__global__ __launch_bounds__(256) void gemm_bf16_wmma(
    const uint16_t* __restrict__ A, const uint16_t* __restrict__ B,
    const float* __restrict__ bias, void* __restrict__ out,
    int M, int N, int K) {
  __shared__ __align__(16) uint16_t As[2][GTM * GTK];  // row-major, stride 32
  __shared__ __align__(16) uint16_t Bt[2][GTN * GTK];  // B transposed: Bt[n][k]

  const int t = threadIdx.x;
  const int w = t >> 5, l = t & 31;
  const int g = l >> 4, r16 = l & 15;
  const int mBase = blockIdx.x * GTM, nBase = blockIdx.y * GTN;
  const int wm = w >> 1, wn = w & 1;  // wave tile origin: (wm*32, wn*32)

  v8f acc[2][2] = {};

  // ---- tile stage: A (async DMA if available), B transposed (sync) ----
  auto stage = [&](uint16_t* AsB, uint16_t* BtB, int kb) {
#ifdef USE_ASYNC_LDS
    {
      // 128x32 bf16 = 512 x 16B transfers; 256 threads x 2
      int e = t;
#pragma unroll
      for (int i = 0; i < 2; ++i, e += 256) {
        int r = e >> 2;          // row 0..127
        int c = (e & 3) * 8;     // col elem 0,8,16,24
        const uint16_t* gp = A + (size_t)(mBase + r) * K + kb + c;
        uint16_t* lp = AsB + r * GTK + c;
        __builtin_amdgcn_global_load_async_to_lds_b128(
            (v4i_g*)(uintptr_t)gp, (v4i_l*)(uintptr_t)lp,
            /*offset=*/0, /*cpol=*/0);
      }
    }
#else
#pragma unroll
    for (int i = 0; i < 8; ++i) {
      int e = t + i * 256;       // dword index 0..2047
      int r = e >> 4, c = e & 15;
      ((uint32_t*)AsB)[e] =
          ((const uint32_t*)(A + (size_t)(mBase + r) * K + kb))[c];
    }
#endif
    // B tile transposed (32x64 -> Bt[64][32]); needs 16-bit transpose
    {
      int k  = t >> 3;           // 0..31
      int n0 = (t & 7) * 8;      // 0..56
      const uint16_t* bp = B + (size_t)(kb + k) * N + nBase + n0;
      uint4 vv = *(const uint4*)bp;  // 8 bf16
      uint16_t tmp[8];
      *(uint4*)tmp = vv;
#pragma unroll
      for (int j = 0; j < 8; ++j) BtB[(n0 + j) * GTK + k] = tmp[j];
    }
  };

  stage(As[0], Bt[0], 0);
  int cur = 0;
  for (int kb = 0; kb < K; kb += GTK) {
    wait_async_lds();
    __syncthreads();
    const int nxt = cur ^ 1;
    if (kb + GTK < K) stage(As[nxt], Bt[nxt], kb + GTK);
    if (kb + 2 * GTK < K) {  // warm L2/WGP$ for the tile after next
      __builtin_prefetch(A + (size_t)(mBase + (t >> 1)) * K + kb + 2 * GTK, 0, 0);
      __builtin_prefetch(B + (size_t)(kb + 2 * GTK + (t >> 3)) * N + nBase +
                             (t & 7) * 8, 0, 0);
    }

    v16bf af[2], bfr[2];
#pragma unroll
    for (int i = 0; i < 2; ++i)
      af[i] = load_frag_a(As[cur] + (size_t)(wm * 32 + i * 16 + r16) * GTK, g);
#pragma unroll
    for (int j = 0; j < 2; ++j)
      bfr[j] = load_frag_b(Bt[cur] + (size_t)(wn * 32 + j * 16 + r16) * GTK, g);

#pragma unroll
    for (int i = 0; i < 2; ++i)
#pragma unroll
      for (int j = 0; j < 2; ++j)
        acc[i][j] = wmma_bf16(af[i], bfr[j], acc[i][j]);
    cur = nxt;
  }

  // --- epilogue: C/D layout: vgpr jj, lane l -> row jj+8*(l>>4), col l&15 ---
#pragma unroll
  for (int i = 0; i < 2; ++i)
#pragma unroll
    for (int j = 0; j < 2; ++j)
#pragma unroll
      for (int jj = 0; jj < 8; ++jj) {
        int row = mBase + wm * 32 + i * 16 + jj + 8 * g;
        int col = nBase + wn * 32 + j * 16 + r16;
        float v = acc[i][j][jj] + bias[col];
        if constexpr (MODE == 0)
          ((float*)out)[(size_t)row * N + col] = v;
        else if constexpr (MODE == 1)
          ((uint16_t*)out)[(size_t)row * N + col] = f2bf(v);
        else
          ((uint16_t*)out)[(size_t)col * M + row] = f2bf(v);
      }
}

// ---------------------------------------------------------------------------
// Attention: per (32-row block, head). Scores kept in 256KB LDS; sparsemax
// via bisection; A*V via WMMA from in-LDS bf16 probabilities.
//   Qp, Kp: [2048 x 1024] bf16 row-major (head h = cols h*64..h*64+63)
//   Vt:     [1024 x 2048] bf16 (Vt[n][s] = V[s][n])
//   Ctx:    [2048 x 1024] bf16 row-major output (already in mix=True layout)
// ---------------------------------------------------------------------------
#define DMODEL 1024
#define SEQ    2048

__global__ __launch_bounds__(256) void hopfield_attn_wmma(
    const uint16_t* __restrict__ Qp, const uint16_t* __restrict__ Kp,
    const uint16_t* __restrict__ Vt, uint16_t* __restrict__ Ctx) {
  extern __shared__ char smem[];          // 32*2048 fp32 = 256KB
  float* sc = (float*)smem;               // scores, row stride SEQ

  const int t = threadIdx.x;
  const int w = t >> 5, l = t & 31;
  const int g = l >> 4, r16 = l & 15;
  const int h = blockIdx.y;
  const int lBlk = blockIdx.x * 32;
  const int hc = h * 64;

  // ---- preload Q fragments: 2 row tiles x 2 k-steps (DK=64) ----
  v16bf qf[2][2];
#pragma unroll
  for (int mt = 0; mt < 2; ++mt)
#pragma unroll
    for (int ks = 0; ks < 2; ++ks)
      qf[mt][ks] = load_frag_a(
          Qp + (size_t)(lBlk + mt * 16 + r16) * DMODEL + hc + ks * 32, g);

  // ---- phase 1: scores = Q K^T * 1/sqrt(64) into LDS ----
  const float scale = 0.125f;
  for (int c = w; c < SEQ / 16; c += 8) {   // 128 chunks of 16 keys, 16/wave
    int sBase = c * 16;
    if (c + 8 < SEQ / 16)  // prefetch next chunk's K rows
      __builtin_prefetch(Kp + (size_t)(sBase + 128 + r16) * DMODEL + hc, 0, 0);
    v16bf kf[2];
#pragma unroll
    for (int ks = 0; ks < 2; ++ks)
      kf[ks] = load_frag_b(Kp + (size_t)(sBase + r16) * DMODEL + hc + ks * 32, g);
    v8f a0 = {}, a1 = {};
    a0 = wmma_bf16(qf[0][0], kf[0], a0);
    a0 = wmma_bf16(qf[0][1], kf[1], a0);
    a1 = wmma_bf16(qf[1][0], kf[0], a1);
    a1 = wmma_bf16(qf[1][1], kf[1], a1);
#pragma unroll
    for (int jj = 0; jj < 8; ++jj) {
      sc[(size_t)(jj + 8 * g) * SEQ + sBase + r16]      = a0[jj] * scale;
      sc[(size_t)(16 + jj + 8 * g) * SEQ + sBase + r16] = a1[jj] * scale;
    }
  }
  __syncthreads();

  // ---- phase 2: sparsemax per row (bisection for tau), wave w: rows 4w..4w+3
  for (int rr = 0; rr < 4; ++rr) {
    int r = w * 4 + rr;
    float z[64];
    float mx = -1e30f;
#pragma unroll
    for (int j = 0; j < 64; ++j) {
      z[j] = sc[(size_t)r * SEQ + l + 32 * j];
      mx = fmaxf(mx, z[j]);
    }
#pragma unroll
    for (int off = 16; off > 0; off >>= 1)
      mx = fmaxf(mx, __shfl_xor(mx, off, 32));
    // tau in [mx-1, mx]; sum(max(z-tau,0)) is monotone decreasing in tau
    float lo = mx - 1.0f, hi = mx;
    for (int it = 0; it < 32; ++it) {
      float mid = 0.5f * (lo + hi);
      float s = 0.0f;
#pragma unroll
      for (int j = 0; j < 64; ++j) s += fmaxf(z[j] - mid, 0.0f);
#pragma unroll
      for (int off = 16; off > 0; off >>= 1) s += __shfl_xor(s, off, 32);
      if (s > 1.0f) lo = mid; else hi = mid;
    }
    float tau = 0.5f * (lo + hi);
    // write probabilities as bf16 into the first half of this row's region
    uint16_t* arow = (uint16_t*)smem + (size_t)r * (SEQ * 2);  // byte off r*8192
#pragma unroll
    for (int j = 0; j < 64; ++j)
      arow[l + 32 * j] = f2bf(fmaxf(z[j] - tau, 0.0f));
  }
  __syncthreads();

  // ---- phase 3: out = A[32xS] * V[Sx64]; wave w -> tile (w&1, w>>1) ----
  const int mt = w & 1, nt = w >> 1;
  v8f acc = {};
#pragma unroll 2
  for (int ks = 0; ks < SEQ / 32; ++ks) {
    const uint16_t* ap =
        (const uint16_t*)smem + (size_t)(mt * 16 + r16) * (SEQ * 2) + ks * 32;
    v16bf afr = load_frag_a(ap, g);
    const uint16_t* vp = Vt + (size_t)(hc + nt * 16 + r16) * SEQ + ks * 32;
    v16bf bfr = load_frag_b(vp, g);
    acc = wmma_bf16(afr, bfr, acc);
  }
#pragma unroll
  for (int jj = 0; jj < 8; ++jj) {
    int row = lBlk + mt * 16 + jj + 8 * g;
    int col = hc + nt * 16 + r16;
    Ctx[(size_t)row * DMODEL + col] = f2bf(acc[jj]);
  }
}

// ---------------------------------------------------------------------------
// Launch
// ---------------------------------------------------------------------------
extern "C" void kernel_launch(void* const* d_in, const int* in_sizes, int n_in,
                              void* d_out, int out_size, void* d_ws,
                              size_t ws_size, hipStream_t stream) {
  const float* queries = (const float*)d_in[0];
  const float* keys    = (const float*)d_in[1];
  // d_in[2] (values) is UNUSED by the reference
  const float* Wq = (const float*)d_in[3];
  const float* bq = (const float*)d_in[4];
  const float* Wk = (const float*)d_in[5];
  const float* bk = (const float*)d_in[6];
  const float* Wv = (const float*)d_in[7];
  const float* bv = (const float*)d_in[8];
  const float* Wo = (const float*)d_in[9];
  const float* bo = (const float*)d_in[10];

  const int L = 2048, D = 1024;
  char* ws = (char*)d_ws;
  const size_t MB = 1ull << 20;
  uint16_t* qbf = (uint16_t*)(ws + 0 * MB);   // 4MB
  uint16_t* kbf = (uint16_t*)(ws + 4 * MB);   // 4MB
  uint16_t* wqb = (uint16_t*)(ws + 8 * MB);   // 2MB
  uint16_t* wkb = (uint16_t*)(ws + 10 * MB);  // 2MB
  uint16_t* wvb = (uint16_t*)(ws + 12 * MB);  // 2MB
  uint16_t* wob = (uint16_t*)(ws + 14 * MB);  // 2MB
  uint16_t* Qp  = (uint16_t*)(ws + 16 * MB);  // 4MB
  uint16_t* Kp  = (uint16_t*)(ws + 20 * MB);  // 4MB
  uint16_t* Vt  = (uint16_t*)(ws + 24 * MB);  // 4MB  (transposed V)
  uint16_t* Ctx = (uint16_t*)(ws + 28 * MB);  // 4MB

  const int nAct = L * D;      // 2M elems
  const int nW   = D * D;      // 1M elems
  cvt_f32_bf16<<<(nAct + 255) / 256, 256, 0, stream>>>(queries, qbf, nAct);
  cvt_f32_bf16<<<(nAct + 255) / 256, 256, 0, stream>>>(keys, kbf, nAct);
  cvt_f32_bf16<<<(nW + 255) / 256, 256, 0, stream>>>(Wq, wqb, nW);
  cvt_f32_bf16<<<(nW + 255) / 256, 256, 0, stream>>>(Wk, wkb, nW);
  cvt_f32_bf16<<<(nW + 255) / 256, 256, 0, stream>>>(Wv, wvb, nW);
  cvt_f32_bf16<<<(nW + 255) / 256, 256, 0, stream>>>(Wo, wob, nW);

  dim3 gg(L / GTM, D / GTN);  // (16, 16)
  // Qp = queries*Wq + bq  (bf16 out)
  gemm_bf16_wmma<1><<<gg, 256, 0, stream>>>(qbf, wqb, bq, Qp, L, D, D);
  // Kp = keys*Wk + bk  (bf16 out)
  gemm_bf16_wmma<1><<<gg, 256, 0, stream>>>(kbf, wkb, bk, Kp, L, D, D);
  // Vt = (Kp*Wv + bv)^T  (bf16 transposed out)
  gemm_bf16_wmma<2><<<gg, 256, 0, stream>>>(Kp, wvb, bv, Vt, L, D, D);

  // attention + sparsemax: 64 row-blocks x 16 heads, 256KB dynamic LDS
  hopfield_attn_wmma<<<dim3(L / 32, 16), 256, 32 * SEQ * sizeof(float),
                       stream>>>(Qp, Kp, Vt, Ctx);

  // out = Ctx*Wo + bo  (f32 out)
  gemm_bf16_wmma<0><<<gg, 256, 0, stream>>>(Ctx, wob, bo, d_out, L, D, D);
}
